// RecurrentGCN1_28853590295018
// MI455X (gfx1250) — matrix operand order repaired and checked by reading, not compiled, not run on hardware
//
#include <hip/hip_runtime.h>
#include <hip/hip_bf16.h>
#include <math.h>

// ---------------------------------------------------------------------------
// Types for CDNA5 WMMA (wave32): v_wmma_f32_16x16x32_f16
// ---------------------------------------------------------------------------
typedef _Float16 half_t;
typedef __attribute__((ext_vector_type(16))) _Float16 v16h;
typedef __attribute__((ext_vector_type(8)))  float    v8f;

#define WMMA(A, B, C) \
    __builtin_amdgcn_wmma_f32_16x16x32_f16(false, (A), false, (B), (short)0, (C), false, false)

#define LOG2E 1.44269504088896340736f

__device__ __forceinline__ float fast_exp2(float x) { return __builtin_amdgcn_exp2f(x); }
__device__ __forceinline__ float fast_rcp(float x)  { return __builtin_amdgcn_rcpf(x); }
// sigmoid(z) = 1/(1+e^-z); saturates cleanly: z->-inf => rcp(inf)=0, z->+inf => rcp(1)=1
__device__ __forceinline__ float fast_sigmoid(float z) {
    return fast_rcp(1.0f + fast_exp2(-z * LOG2E));
}
// tanh(a) = 1 - 2/(e^{2a}+1); a->+inf => 1-0=1, a->-inf => 1-2=-1 (no NaN)
__device__ __forceinline__ float fast_tanh(float a) {
    return 1.0f - 2.0f * fast_rcp(fast_exp2(2.0f * LOG2E * a) + 1.0f);
}

__device__ __forceinline__ v16h zero_v16h() {
    v16h z;
#pragma unroll
    for (int i = 0; i < 16; ++i) z[i] = (_Float16)0.0f;
    return z;
}
__device__ __forceinline__ v8f zero_v8f() {
    v8f z;
#pragma unroll
    for (int i = 0; i < 8; ++i) z[i] = 0.0f;
    return z;
}

// A fragment: 16x32 f16 tile, row-major source, leading dim ld.
// Contiguous k order used here (A and B match): lanes 0-15 hold row M=lane, K = 0..15;
//             lanes 16-31 hold row M=lane-16, K = 16..31.
__device__ __forceinline__ v16h load_A(const half_t* __restrict__ src, int ld) {
    const int lane = threadIdx.x & 31;
    const int m  = lane & 15;
    const int kb = (lane >> 4) << 4;           // 0 or 16
    const half_t* p0 = src + m * ld + kb;      // K = kb..kb+7
    const half_t* p1 = p0 + 8;                 // K = kb+8..kb+15
    v16h a;
#pragma unroll
    for (int i = 0; i < 8; ++i) { a[i] = p0[i]; a[8 + i] = p1[i]; }
    return a;
}

// B fragment from B^T storage (row-major [n][k], leading dim ldt).
// Contiguous k order (matches load_A): lanes 0-15 hold col N=lane, K=0..15; lanes 16-31 col N=lane-16, K=16..31.
__device__ __forceinline__ v16h load_BT(const half_t* __restrict__ bt, int ldt) {
    const int lane = threadIdx.x & 31;
    const int n  = lane & 15;
    const int kb = (lane >> 4) << 4;           // 0 or 16
    const half_t* p = bt + n * ldt + kb;
    v16h b;
#pragma unroll
    for (int i = 0; i < 16; ++i) b[i] = p[i];
    return b;
}

// C layout: col = lane&15, row(j) = j + 8*(lane>>4)

// ---------------------------------------------------------------------------
// Problem constants
// ---------------------------------------------------------------------------
#define N_NODES 4096
#define F_IN    64
#define HD      128
#define NH      8
#define DH      16

// ---------------------------------------------------------------------------
// Kernel 0: precompute f16 operands. Wz_eff/Wh_eff folded and transposed to
// [n][k]; Linear weights ([out,in]) are already B^T layout -> plain convert.
// ---------------------------------------------------------------------------
__global__ void k_prep(const float* __restrict__ x,
                       const float* __restrict__ Wz, const float* __restrict__ Wh,
                       const float* __restrict__ ipw, const float* __restrict__ opw,
                       const float* __restrict__ fc1w,
                       half_t* __restrict__ x16,
                       half_t* __restrict__ WzT, half_t* __restrict__ WhT,
                       half_t* __restrict__ W3,  half_t* __restrict__ Wout,
                       half_t* __restrict__ Wfc1) {
    const int i = blockIdx.x * blockDim.x + threadIdx.x;
    if (i < N_NODES * F_IN) x16[i] = (half_t)x[i];
    if (i < HD * F_IN) {                    // WzT/WhT: [n=128][k=64]
        const int n = i >> 6, k = i & 63;   // W shape (2, 192, 128); only rows k<64 matter
        WzT[i] = (half_t)(Wz[k * HD + n] + Wz[192 * HD + k * HD + n]);
        WhT[i] = (half_t)(Wh[k * HD + n] + Wh[192 * HD + k * HD + n]);
    }
    if (i < 3 * HD * HD) W3[i] = (half_t)ipw[i];
    if (i < HD * HD) { Wout[i] = (half_t)opw[i]; Wfc1[i] = (half_t)fc1w[i]; }
}

// ---------------------------------------------------------------------------
// Kernel 1: gates.  h = (1 - sigmoid(x@WzT^T + bz)) * tanh(x@WhT^T + bh)
// One wave per 16-row tile; 8 column tiles, K=64 -> 2 wmma per accumulator.
// ---------------------------------------------------------------------------
__global__ void __launch_bounds__(128) k_gates(const half_t* __restrict__ x16,
                                               const half_t* __restrict__ WzT,
                                               const half_t* __restrict__ WhT,
                                               const float* __restrict__ bz,
                                               const float* __restrict__ bh,
                                               float* __restrict__ h32,
                                               half_t* __restrict__ h16) {
    const int wave = threadIdx.x >> 5, lane = threadIdx.x & 31;
    const int rt = blockIdx.x * 4 + wave;            // 0..255
    const int hi = lane >> 4, l16 = lane & 15;
    const half_t* A = x16 + rt * 16 * F_IN;
    const v16h a0 = load_A(A, F_IN);
    const v16h a1 = load_A(A + 32, F_IN);
    for (int ct = 0; ct < 8; ++ct) {
        const half_t* Bz = WzT + ct * 16 * F_IN;
        const half_t* Bh = WhT + ct * 16 * F_IN;
        v8f accz = zero_v8f(), acch = zero_v8f();
        accz = WMMA(a0, load_BT(Bz, F_IN), accz);
        accz = WMMA(a1, load_BT(Bz + 32, F_IN), accz);
        acch = WMMA(a0, load_BT(Bh, F_IN), acch);
        acch = WMMA(a1, load_BT(Bh + 32, F_IN), acch);
        const int col = ct * 16 + l16;
        const float bzv = bz[col], bhv = bh[col];
#pragma unroll
        for (int j = 0; j < 8; ++j) {
            const int row = rt * 16 + j + 8 * hi;
            const float z  = fast_sigmoid(accz[j] + bzv);
            const float th = fast_tanh(acch[j] + bhv);
            const float hv = (1.0f - z) * th;
            h32[row * HD + col] = hv;
            h16[row * HD + col] = (half_t)hv;
        }
    }
}

// ---------------------------------------------------------------------------
// Kernel 2: qkv projection. qkv = h @ in_proj_w^T + b ; scatter per head:
//   Q[head][row][d] pre-scaled by (1/sqrt(DH)) * log2(e) so the online
//   softmax can run entirely in base-2 (v_exp_f32), K[head][row][d],
//   VT[head][d][row]  (transposed so P@V's B^T fragment is contiguous)
// ---------------------------------------------------------------------------
__global__ void __launch_bounds__(128) k_qkv(const half_t* __restrict__ h16,
                                             const half_t* __restrict__ W3,
                                             const float* __restrict__ ipb,
                                             half_t* __restrict__ Q,
                                             half_t* __restrict__ Km,
                                             half_t* __restrict__ VT) {
    const int wave = threadIdx.x >> 5, lane = threadIdx.x & 31;
    const int rt = blockIdx.x * 4 + wave;            // 0..255
    const int hi = lane >> 4, l16 = lane & 15;
    const half_t* A = h16 + rt * 16 * HD;
    const v16h a0 = load_A(A, HD),       a1 = load_A(A + 32, HD);
    const v16h a2 = load_A(A + 64, HD),  a3 = load_A(A + 96, HD);
    const float qscale = 0.25f * LOG2E;              // 1/sqrt(16) * log2(e)
    for (int ct = 0; ct < 24; ++ct) {
        const half_t* B = W3 + ct * 16 * HD;
        v8f acc = zero_v8f();
        acc = WMMA(a0, load_BT(B, HD), acc);
        acc = WMMA(a1, load_BT(B + 32, HD), acc);
        acc = WMMA(a2, load_BT(B + 64, HD), acc);
        acc = WMMA(a3, load_BT(B + 96, HD), acc);
        const int n = ct * 16 + l16;                 // 0..383
        const float bias = ipb[n];
        const int sec = n >> 7, c = n & 127, head = c >> 4, d = c & 15;
#pragma unroll
        for (int j = 0; j < 8; ++j) {
            const int row = rt * 16 + j + 8 * hi;
            const float v = acc[j] + bias;
            if (sec == 0)      Q [(head * N_NODES + row) * DH + d] = (half_t)(v * qscale);
            else if (sec == 1) Km[(head * N_NODES + row) * DH + d] = (half_t)v;
            else               VT[(head * DH + d) * N_NODES + row] = (half_t)v;
        }
    }
}

// ---------------------------------------------------------------------------
// Kernel 3: flash attention. One wave per (head, 16-query tile); online
// softmax (base-2, single v_exp_f32 per element) over 32-key chunks:
// 2 score WMMAs (Q zero-padded K 16..31), P transposed through per-wave LDS,
// 1 PV WMMA per chunk. Next chunk's K/V lines prefetched (global_prefetch_b8).
// ---------------------------------------------------------------------------
__global__ void __launch_bounds__(128, 1) k_attn(const half_t* __restrict__ Q,
                                                 const half_t* __restrict__ Km,
                                                 const half_t* __restrict__ VT,
                                                 half_t* __restrict__ ctx16) {
    __shared__ half_t plds[4][16 * 32];
    const int wave = threadIdx.x >> 5, lane = threadIdx.x & 31;
    const int gw = blockIdx.x * 4 + wave;            // 0..2047
    const int head = gw >> 8, qt = gw & 255;
    const int hi = lane >> 4, l16 = lane & 15;

    // Q fragment: 16x16 real data, K=16..31 zero padded
    v16h qf = zero_v16h();
    if (hi == 0) {
        const half_t* qp = Q + (head * N_NODES + qt * 16 + l16) * DH;
#pragma unroll
        for (int i = 0; i < 16; ++i) qf[i] = qp[i];   // lanes 16-31 (K 16..31) stay 0
    }

    v8f ctx = zero_v8f();
    float rm[8], rl[8];
#pragma unroll
    for (int j = 0; j < 8; ++j) { rm[j] = -3.0e38f; rl[j] = 0.0f; }
    half_t* pw = plds[wave];
    const v8f zc = zero_v8f();

    const half_t* kbase = Km + head * N_NODES * DH;        // [key][d]
    const half_t* vbase = VT + (head * DH + l16) * N_NODES; // row d=l16, keys contiguous

    for (int kt = 0; kt < N_NODES / 32; ++kt) {
        const int key0 = kt * 32;
        // Prefetch next chunk's K rows and V segment into cache while we do
        // softmax VALU work (lowers to global_prefetch_b8).
        if (kt + 1 < N_NODES / 32) {
            __builtin_prefetch(kbase + (key0 + 32 + lane) * DH, 0, 3);
            __builtin_prefetch(vbase + key0 + 32 + (hi * 16), 0, 3);
        }
        // B^T fragments for scores: row-major K matrix is exactly B^T layout.
        // Lanes 16-31 hold features 16..31 of the padded K-dim -> zero.
        v16h bk0 = zero_v16h(), bk1 = zero_v16h();
        if (lane < 16) {
            const half_t* kp0 = kbase + (key0 + l16) * DH;
            const half_t* kp1 = kp0 + 16 * DH;
#pragma unroll
            for (int i = 0; i < 16; ++i) { bk0[i] = kp0[i]; bk1[i] = kp1[i]; }
        }
        v8f s0 = WMMA(qf, bk0, zc);     // scores already in log2 domain
        v8f s1 = WMMA(qf, bk1, zc);

        // online softmax; rows live in 16-lane groups, shfl_xor masks <= 8 stay in-group
#pragma unroll
        for (int j = 0; j < 8; ++j) {
            float cm = fmaxf(s0[j], s1[j]);
            cm = fmaxf(cm, __shfl_xor(cm, 1));
            cm = fmaxf(cm, __shfl_xor(cm, 2));
            cm = fmaxf(cm, __shfl_xor(cm, 4));
            cm = fmaxf(cm, __shfl_xor(cm, 8));
            const float nm   = fmaxf(rm[j], cm);
            const float corr = fast_exp2(rm[j] - nm);
            const float p0   = fast_exp2(s0[j] - nm);
            const float p1   = fast_exp2(s1[j] - nm);
            float rs = p0 + p1;
            rs += __shfl_xor(rs, 1);
            rs += __shfl_xor(rs, 2);
            rs += __shfl_xor(rs, 4);
            rs += __shfl_xor(rs, 8);
            rl[j] = rl[j] * corr + rs;
            rm[j] = nm;
            ctx[j] *= corr;
            const int r = j + 8 * hi;                 // C-layout row
            pw[r * 32 + l16]      = (half_t)p0;       // keys key0..key0+15
            pw[r * 32 + 16 + l16] = (half_t)p1;       // keys key0+16..+31
        }
        // cross-lane LDS write -> read: drain DS counter (CDNA5 split waits)
        asm volatile("s_wait_dscnt 0" ::: "memory");

        // P as A-fragment (16 rows x 32 keys) from LDS
        v16h pf;
        {
            const int kb = hi * 16;
            const half_t* pp = pw + l16 * 32 + kb;
#pragma unroll
            for (int i = 0; i < 8; ++i) { pf[i] = pp[i]; pf[8 + i] = pp[8 + i]; }
        }
        // V^T as B^T fragment: contiguous keys per (head, d) row
        v16h vf;
        {
            const int ko = hi * 16;
            const half_t* vp = vbase + key0 + ko;
#pragma unroll
            for (int i = 0; i < 16; ++i) vf[i] = vp[i];
        }
        ctx = WMMA(pf, vf, ctx);
    }

#pragma unroll
    for (int j = 0; j < 8; ++j) {
        const float v = ctx[j] * fast_rcp(rl[j]);
        const int row = qt * 16 + j + 8 * hi;
        ctx16[row * HD + head * DH + l16] = (half_t)v;
    }
}

// ---------------------------------------------------------------------------
// Kernel 4: out_proj + residual:  h2 = h + ctx @ out_proj_w^T + b
// ---------------------------------------------------------------------------
__global__ void __launch_bounds__(128) k_oproj(const half_t* __restrict__ ctx16,
                                               const half_t* __restrict__ Wout,
                                               const float* __restrict__ opb,
                                               const float* __restrict__ h32,
                                               half_t* __restrict__ h2) {
    const int wave = threadIdx.x >> 5, lane = threadIdx.x & 31;
    const int rt = blockIdx.x * 4 + wave;
    const int hi = lane >> 4, l16 = lane & 15;
    const half_t* A = ctx16 + rt * 16 * HD;
    const v16h a0 = load_A(A, HD),      a1 = load_A(A + 32, HD);
    const v16h a2 = load_A(A + 64, HD), a3 = load_A(A + 96, HD);
    for (int ct = 0; ct < 8; ++ct) {
        const half_t* B = Wout + ct * 16 * HD;
        v8f acc = zero_v8f();
        acc = WMMA(a0, load_BT(B, HD), acc);
        acc = WMMA(a1, load_BT(B + 32, HD), acc);
        acc = WMMA(a2, load_BT(B + 64, HD), acc);
        acc = WMMA(a3, load_BT(B + 96, HD), acc);
        const int col = ct * 16 + l16;
        const float bias = opb[col];
#pragma unroll
        for (int j = 0; j < 8; ++j) {
            const int row = rt * 16 + j + 8 * hi;
            h2[row * HD + col] = (half_t)(acc[j] + bias + h32[row * HD + col]);
        }
    }
}

// ---------------------------------------------------------------------------
// Kernel 5: fc1 + ReLU + fc2 fused.  out[row] = relu(h2@fc1^T + b1) . fc2 + b2
// ---------------------------------------------------------------------------
__global__ void __launch_bounds__(128) k_fc(const half_t* __restrict__ h2,
                                            const half_t* __restrict__ Wfc1,
                                            const float* __restrict__ b1,
                                            const float* __restrict__ fc2w,
                                            const float* __restrict__ fc2b,
                                            float* __restrict__ out) {
    const int wave = threadIdx.x >> 5, lane = threadIdx.x & 31;
    const int rt = blockIdx.x * 4 + wave;
    const int hi = lane >> 4, l16 = lane & 15;
    const half_t* A = h2 + rt * 16 * HD;
    const v16h a0 = load_A(A, HD),      a1 = load_A(A + 32, HD);
    const v16h a2 = load_A(A + 64, HD), a3 = load_A(A + 96, HD);
    float part[8];
#pragma unroll
    for (int j = 0; j < 8; ++j) part[j] = 0.0f;
    for (int ct = 0; ct < 8; ++ct) {
        const half_t* B = Wfc1 + ct * 16 * HD;
        v8f acc = zero_v8f();
        acc = WMMA(a0, load_BT(B, HD), acc);
        acc = WMMA(a1, load_BT(B + 32, HD), acc);
        acc = WMMA(a2, load_BT(B + 64, HD), acc);
        acc = WMMA(a3, load_BT(B + 96, HD), acc);
        const int col = ct * 16 + l16;
        const float w2 = fc2w[col], bb = b1[col];
#pragma unroll
        for (int j = 0; j < 8; ++j) part[j] += fmaxf(acc[j] + bb, 0.0f) * w2;
    }
#pragma unroll
    for (int j = 0; j < 8; ++j) {
        float s = part[j];
        s += __shfl_xor(s, 1);
        s += __shfl_xor(s, 2);
        s += __shfl_xor(s, 4);
        s += __shfl_xor(s, 8);
        if (l16 == 0) out[rt * 16 + j + 8 * hi] = s + fc2b[0];
    }
}

// ---------------------------------------------------------------------------
// Host-side launch
// ---------------------------------------------------------------------------
extern "C" void kernel_launch(void* const* d_in, const int* in_sizes, int n_in,
                              void* d_out, int out_size, void* d_ws, size_t ws_size,
                              hipStream_t stream) {
    const float* x    = (const float*)d_in[0];
    // d_in[1] = edge_index (mathematically inert for K=1 DConv)
    const float* Wz   = (const float*)d_in[2];
    const float* bz   = (const float*)d_in[3];
    const float* Wh   = (const float*)d_in[6];
    const float* bh   = (const float*)d_in[7];
    const float* ipw  = (const float*)d_in[8];
    const float* ipb  = (const float*)d_in[9];
    const float* opw  = (const float*)d_in[10];
    const float* opb  = (const float*)d_in[11];
    const float* fc1w = (const float*)d_in[12];
    const float* fc1b = (const float*)d_in[13];
    const float* fc2w = (const float*)d_in[14];
    const float* fc2b = (const float*)d_in[15];
    float* out = (float*)d_out;

    // Workspace carve (~8.7 MB total)
    half_t* x16  = (half_t*)d_ws;               // 4096*64
    half_t* WzT  = x16  + N_NODES * F_IN;       // 128*64
    half_t* WhT  = WzT  + HD * F_IN;
    half_t* W3   = WhT  + HD * F_IN;            // 384*128
    half_t* Wout = W3   + 3 * HD * HD;          // 128*128
    half_t* Wfc1 = Wout + HD * HD;              // 128*128
    float*  h32  = (float*)(Wfc1 + HD * HD);    // 4096*128 f32
    half_t* h16  = (half_t*)(h32 + N_NODES * HD);
    half_t* Q    = h16  + N_NODES * HD;         // [NH][N][DH]
    half_t* Km   = Q    + N_NODES * HD;         // [NH][N][DH]
    half_t* VT   = Km   + N_NODES * HD;         // [NH][DH][N]
    half_t* ctx  = VT   + N_NODES * HD;         // [N][HD]
    half_t* h2   = ctx  + N_NODES * HD;         // [N][HD]

    k_prep <<<(N_NODES * F_IN) / 256, 256, 0, stream>>>(x, Wz, Wh, ipw, opw, fc1w,
                                                        x16, WzT, WhT, W3, Wout, Wfc1);
    k_gates<<<64,  128, 0, stream>>>(x16, WzT, WhT, bz, bh, h32, h16);
    k_qkv  <<<64,  128, 0, stream>>>(h16, W3, ipb, Q, Km, VT);
    k_attn <<<512, 128, 0, stream>>>(Q, Km, VT, ctx);
    k_oproj<<<64,  128, 0, stream>>>(ctx, Wout, opb, h32, h2);
    k_fc   <<<64,  128, 0, stream>>>(h2, Wfc1, fc1b, fc2w, fc2b, out);
}
